// HSLPart2_47278999994503
// MI455X (gfx1250) — matrix operand order repaired and checked by reading, not compiled, not run on hardware
//
#include <hip/hip_runtime.h>

typedef __attribute__((ext_vector_type(16))) _Float16 v16h;
typedef __attribute__((ext_vector_type(8)))  float    v8f;
typedef __attribute__((ext_vector_type(4)))  _Float16 h4;

#define NN      4096
#define MM      4096
#define DD      128
#define NC      4
#define NNZ     262144
#define NUMADD  26214
#define TOTAL   (NN * MM)

// ---------------- workspace layout (bytes) ----------------
#define WS_EX      0u                      // eX: M*D f32            (2 MB)
#define WS_CNT     2097152u                // counts: M u32          (16 KB)
#define WS_NF      2113536u                // node_fc: N*512 f16     (4 MB)
#define WS_EF      6307840u                // edge_fc: M*512 f16     (4 MB)
#define WS_H0      10502144u               // hist0: 2048 u32
#define WS_H1      10510336u               // hist1: 2048 u32
#define WS_H2      10518528u               // hist2: 1024 u32
#define WS_STATE   10522624u               // state: 4 u32 {prefix, k_rem, thresh_key, bnd_cnt}

__device__ __forceinline__ unsigned fkey(float f) {
    unsigned u = __float_as_uint(f);
    return (u & 0x80000000u) ? ~u : (u | 0x80000000u);
}

// ---------------- 0: zero scratch + init select state ----------------
__global__ void hsl_zero_kernel(unsigned* ws_u32, unsigned* hist_u32, unsigned* state) {
    int idx = blockIdx.x * blockDim.x + threadIdx.x;
    // eX + counts region: (2097152 + 16384)/4 words
    for (int i = idx; i < 528384; i += gridDim.x * blockDim.x) ws_u32[i] = 0u;
    // hist0 + hist1 + hist2: (8192+8192+4096)/4 words
    if (idx < 5120) hist_u32[idx] = 0u;
    if (idx == 0) {
        state[0] = 0u;          // prefix
        state[1] = NUMADD;      // k remaining
        state[2] = 0u;          // threshold key
        state[3] = 0u;          // boundary counter
    }
}

// ---------------- 1: scatter-sum X[V] into eX[E], count per edge ----------------
__global__ __launch_bounds__(256) void hsl_scatter_kernel(const float* __restrict__ X,
                                                          const int* __restrict__ V,
                                                          const int* __restrict__ E,
                                                          float* __restrict__ eX,
                                                          unsigned* __restrict__ counts) {
    int gid  = blockIdx.x * blockDim.x + threadIdx.x;
    int nz   = gid >> 5;           // one wave per (v,e) pair
    int lane = gid & 31;
    if (nz >= NNZ) return;
    int v = V[nz], e = E[nz];
    const float4 x = *(const float4*)(X + v * DD + lane * 4);
    float* dst = eX + e * DD + lane * 4;
    atomicAdd(dst + 0, x.x);
    atomicAdd(dst + 1, x.y);
    atomicAdd(dst + 2, x.z);
    atomicAdd(dst + 3, x.w);
    if (lane == 0) atomicAdd(counts + e, 1u);
}

// ---------------- 2: per-channel L2-normalized features -> f16, flattened (c,d) ----------------
// divide by N_C folded into the row scale so S = nf @ ef^T directly.
__device__ __forceinline__ void row_features(float x0, float x1, float x2, float x3,
                                             const float* __restrict__ cw,
                                             _Float16* __restrict__ out_row,
                                             int lane, float chan_scale) {
    #pragma unroll
    for (int c = 0; c < NC; ++c) {
        const float4 w = *(const float4*)(cw + c * DD + lane * 4);
        float y0 = x0 * w.x, y1 = x1 * w.y, y2 = x2 * w.z, y3 = x3 * w.w;
        float ss = y0 * y0 + y1 * y1 + y2 * y2 + y3 * y3;
        #pragma unroll
        for (int off = 16; off > 0; off >>= 1) ss += __shfl_xor(ss, off, 32);
        float inv = chan_scale / fmaxf(sqrtf(ss), 1e-12f);
        h4 o;
        o.x = (_Float16)(y0 * inv);
        o.y = (_Float16)(y1 * inv);
        o.z = (_Float16)(y2 * inv);
        o.w = (_Float16)(y3 * inv);
        *(h4*)(out_row + c * DD + lane * 4) = o;
    }
}

__global__ __launch_bounds__(256) void hsl_nodefc_kernel(const float* __restrict__ X,
                                                         const float* __restrict__ cw,
                                                         _Float16* __restrict__ nf) {
    int lane = threadIdx.x & 31, wave = threadIdx.x >> 5;
    int n = blockIdx.x * 8 + wave;
    const float4 x = *(const float4*)(X + n * DD + lane * 4);
    // fold the 1/N_C of the final mean into the node side (sqrt split keeps f16 range healthy)
    row_features(x.x, x.y, x.z, x.w, cw, nf + (size_t)n * 512, lane, 0.5f);
}

__global__ __launch_bounds__(256) void hsl_edgefc_kernel(const float* __restrict__ eX,
                                                         const unsigned* __restrict__ counts,
                                                         const float* __restrict__ cw,
                                                         _Float16* __restrict__ ef) {
    int lane = threadIdx.x & 31, wave = threadIdx.x >> 5;
    int m = blockIdx.x * 8 + wave;
    float invc = 1.0f / fmaxf((float)counts[m], 1.0f);
    const float4 x = *(const float4*)(eX + m * DD + lane * 4);
    row_features(x.x * invc, x.y * invc, x.z * invc, x.w * invc, cw,
                 ef + (size_t)m * 512, lane, 0.5f);
}

// ---------------- 3: S = nf[N,512] @ ef[M,512]^T via v_wmma_f32_16x16x32_f16 ----------------
union FragAB { v16h v; unsigned u[8]; };

__global__ __launch_bounds__(256) void hsl_gemm_kernel(const _Float16* __restrict__ nf,
                                                       const _Float16* __restrict__ ef,
                                                       float* __restrict__ S) {
    __shared__ __align__(16) _Float16 As[128][40];   // padded: 80B stride
    __shared__ __align__(16) _Float16 Bs[128][40];

    const int tid  = threadIdx.x;
    const int lane = tid & 31;
    const int wave = tid >> 5;
    const int wm = wave >> 2;          // 0..1  (64 rows each)
    const int wn = wave & 3;           // 0..3  (32 cols each)
    const int hh = lane >> 4;          // half-wave
    const int nn16 = lane & 15;
    const int rowBase = blockIdx.y * 128;
    const int colBase = blockIdx.x * 128;
    const int lr = tid >> 1;           // 128 rows, 2 threads/row
    const int lk = (tid & 1) * 16;     // 16-halfword chunk

    const v8f vzero = {0.f, 0.f, 0.f, 0.f, 0.f, 0.f, 0.f, 0.f};
    v8f acc[4][2];
    #pragma unroll
    for (int i = 0; i < 4; ++i)
        #pragma unroll
        for (int j = 0; j < 2; ++j) acc[i][j] = vzero;

    for (int kt = 0; kt < 16; ++kt) {      // K = 512, step 32
        __syncthreads();
        *(uint4*)(&As[lr][lk]) =
            *(const uint4*)(nf + (size_t)(rowBase + lr) * 512 + kt * 32 + lk);
        *(uint4*)(&Bs[lr][lk]) =
            *(const uint4*)(ef + (size_t)(colBase + lr) * 512 + kt * 32 + lk);
        __syncthreads();

        FragAB afrag[4], bfrag[2];
        #pragma unroll
        for (int i = 0; i < 4; ++i) {
            const int m = wm * 64 + i * 16 + nn16;
            #pragma unroll
            for (int v = 0; v < 8; ++v) {
                const int k0 = ((v >> 2) << 4) + (hh << 3) + ((v & 3) << 1);
                afrag[i].u[v] = *(const unsigned*)(&As[m][k0]);
            }
        }
        #pragma unroll
        for (int j = 0; j < 2; ++j) {
            const int n = wn * 32 + j * 16 + nn16;
            #pragma unroll
            for (int v = 0; v < 8; ++v) {
                const int k0 = ((v >> 2) << 4) + (hh << 3) + ((v & 3) << 1);
                bfrag[j].u[v] = *(const unsigned*)(&Bs[n][k0]);
            }
        }
        #pragma unroll
        for (int i = 0; i < 4; ++i)
            #pragma unroll
            for (int j = 0; j < 2; ++j)
                acc[i][j] = __builtin_amdgcn_wmma_f32_16x16x32_f16(
                    false, afrag[i].v, false, bfrag[j].v,
                    (short)0, acc[i][j], false, false);
    }

    // C/D layout: VGPR r -> M = r + 8*half, N = lane&15
    #pragma unroll
    for (int i = 0; i < 4; ++i)
        #pragma unroll
        for (int j = 0; j < 2; ++j)
            #pragma unroll
            for (int r = 0; r < 8; ++r) {
                int row = rowBase + wm * 64 + i * 16 + hh * 8 + r;
                int col = colBase + wn * 32 + j * 16 + nn16;
                S[(size_t)row * MM + col] = acc[i][j][r];
            }
}

// ---------------- 4: mask existing incidences ----------------
__global__ void hsl_mask_kernel(const int* __restrict__ V, const int* __restrict__ E,
                                float* __restrict__ S) {
    int i = blockIdx.x * blockDim.x + threadIdx.x;
    if (i >= NNZ) return;
    S[(size_t)V[i] * MM + E[i]] = -1e30f;
}

// ---------------- 5: radix-select (3 passes over float-bit keys) ----------------
__global__ __launch_bounds__(256) void hsl_hist_kernel(const float* __restrict__ S,
                                                       unsigned* __restrict__ hist,
                                                       const unsigned* __restrict__ state,
                                                       int pbits, int dbits) {
    __shared__ unsigned lh[2048];
    const int nb = 1 << dbits;
    for (int i = threadIdx.x; i < nb; i += 256) lh[i] = 0u;
    __syncthreads();
    const unsigned prefix = state[0];
    const int shift = 32 - pbits - dbits;
    const int stride = gridDim.x * blockDim.x;
    for (int idx = blockIdx.x * blockDim.x + threadIdx.x; idx < TOTAL; idx += stride) {
        unsigned key = fkey(S[idx]);
        bool ok = (pbits == 0) || ((key >> (32 - pbits)) == prefix);
        if (ok) atomicAdd(&lh[(key >> shift) & (nb - 1)], 1u);
    }
    __syncthreads();
    for (int i = threadIdx.x; i < nb; i += 256)
        if (lh[i]) atomicAdd(&hist[i], lh[i]);
}

__global__ void hsl_scan_kernel(const unsigned* __restrict__ hist,
                                unsigned* __restrict__ state, int dbits, int final_pass) {
    if (threadIdx.x != 0 || blockIdx.x != 0) return;
    const int nb = 1 << dbits;
    unsigned k = state[1];
    unsigned pre = state[0];
    unsigned cum = 0;
    for (int d = nb - 1; d >= 0; --d) {
        unsigned c = hist[d];
        if (cum + c >= k) {
            state[0] = (pre << dbits) | (unsigned)d;
            state[1] = k - cum;
            break;
        }
        cum += c;
    }
    if (final_pass) state[2] = state[0];
}

// ---------------- 6: fused finalize: enriched_H * gumbel-hard (in place over S) ----------------
__global__ __launch_bounds__(256) void hsl_final_kernel(const float* __restrict__ H,
                                                        const float* __restrict__ imp,
                                                        const float* __restrict__ eps,
                                                        float* __restrict__ S_out,
                                                        unsigned* __restrict__ state) {
    int idx = blockIdx.x * blockDim.x + threadIdx.x;
    if (idx >= TOTAL) return;
    const float s = S_out[idx];
    const unsigned key = fkey(s);
    const unsigned tk = state[2];
    float add = 0.f;
    if (key > tk) {
        add = 1.f;
    } else if (key == tk) {
        if (atomicAdd(&state[3], 1u) < state[1]) add = 1.f;  // boundary quota (exact f32 ties only)
    }
    const float p = imp[idx];
    const float e = eps[idx];
    // hard = (sigmoid(logit/T) > 0.5) == (logit > 0); STE leaves value == hard
    const float logit = __logf(e) - __logf(1.f - e)
                      + __logf(p + 1e-8f) - __logf(1.f - p + 1e-8f);
    const float hard = (logit > 0.f) ? 1.f : 0.f;
    S_out[idx] = (H[idx] + add) * hard;
}

// ---------------- launch ----------------
extern "C" void kernel_launch(void* const* d_in, const int* in_sizes, int n_in,
                              void* d_out, int out_size, void* d_ws, size_t ws_size,
                              hipStream_t stream) {
    const float* X   = (const float*)d_in[0];
    const float* H   = (const float*)d_in[1];
    const int*   V   = (const int*)d_in[2];
    const int*   E   = (const int*)d_in[3];
    const float* imp = (const float*)d_in[4];
    const float* cw  = (const float*)d_in[5];
    const float* eps = (const float*)d_in[6];
    float* S = (float*)d_out;                       // 64 MB: scores, then final output in place

    char* ws = (char*)d_ws;
    float*     eX     = (float*)(ws + WS_EX);
    unsigned*  counts = (unsigned*)(ws + WS_CNT);
    _Float16*  nf     = (_Float16*)(ws + WS_NF);
    _Float16*  ef     = (_Float16*)(ws + WS_EF);
    unsigned*  h0     = (unsigned*)(ws + WS_H0);
    unsigned*  h1     = (unsigned*)(ws + WS_H1);
    unsigned*  h2     = (unsigned*)(ws + WS_H2);
    unsigned*  state  = (unsigned*)(ws + WS_STATE);

    hsl_zero_kernel<<<2064, 256, 0, stream>>>((unsigned*)(ws + WS_EX), h0, state);
    hsl_scatter_kernel<<<NNZ * 32 / 256, 256, 0, stream>>>(X, V, E, eX, counts);
    hsl_nodefc_kernel<<<NN / 8, 256, 0, stream>>>(X, cw, nf);
    hsl_edgefc_kernel<<<MM / 8, 256, 0, stream>>>(eX, counts, cw, ef);

    dim3 ggrid(MM / 128, NN / 128);
    hsl_gemm_kernel<<<ggrid, 256, 0, stream>>>(nf, ef, S);

    hsl_mask_kernel<<<NNZ / 256, 256, 0, stream>>>(V, E, S);

    hsl_hist_kernel<<<2048, 256, 0, stream>>>(S, h0, state, 0, 11);
    hsl_scan_kernel<<<1, 32, 0, stream>>>(h0, state, 11, 0);
    hsl_hist_kernel<<<2048, 256, 0, stream>>>(S, h1, state, 11, 11);
    hsl_scan_kernel<<<1, 32, 0, stream>>>(h1, state, 11, 0);
    hsl_hist_kernel<<<2048, 256, 0, stream>>>(S, h2, state, 22, 10);
    hsl_scan_kernel<<<1, 32, 0, stream>>>(h2, state, 10, 1);

    hsl_final_kernel<<<TOTAL / 256, 256, 0, stream>>>(H, imp, eps, S, state);
}
